// BlockDiT_69715909148723
// MI455X (gfx1250) — matrix operand order repaired and checked by reading, not compiled, not run on hardware
//
#include <hip/hip_runtime.h>
#include <hip/hip_bf16.h>

// ---------------------------------------------------------------------------
// Types for CDNA5 WMMA (wave32): v_wmma_f32_16x16x32_bf16
// ---------------------------------------------------------------------------
typedef __attribute__((ext_vector_type(16))) __bf16 bf16x16;
typedef __attribute__((ext_vector_type(8)))  __bf16 bf16x8;
typedef __attribute__((ext_vector_type(8)))  float  f32x8;

#define DIM   768
#define SEQ   2048
#define NHALF 1024
#define HEADS 12
#define HD    64
#define DFF   3072
#define COND  128
#define ROWS  4096          // 2 * SEQ
#define ADA6  4608          // 6*DIM

__device__ __forceinline__ unsigned short f2bf(float f) {
    unsigned int u = __float_as_uint(f);
    unsigned int r = u + 0x7FFFu + ((u >> 16) & 1u);   // round-to-nearest-even
    return (unsigned short)(r >> 16);
}

// Load a 16x32 bf16 A-fragment from a row-major bf16 (u16) row pointer.
// lane&15 = row within tile, hi = lane>>4 selects the K sub-block (ISA 7.12.2).
__device__ __forceinline__ bf16x16 load_a_frag(const unsigned short* rowptr,
                                               int k0, int hi) {
    bf16x8 lo = *reinterpret_cast<const bf16x8*>(rowptr + k0 + hi * 8);
    bf16x8 hh = *reinterpret_cast<const bf16x8*>(rowptr + k0 + 16 + hi * 8);
    bf16x16 a;
#pragma unroll
    for (int i = 0; i < 8; ++i) { a[i] = lo[i]; a[8 + i] = hh[i]; }
    return a;
}

// ---------------------------------------------------------------------------
// 1) ada = c @ ada_w + ada_b          [2, 4608], K = 128 (tiny)
// ---------------------------------------------------------------------------
__global__ __launch_bounds__(256) void k_ada(const float* __restrict__ c,
                                             const float* __restrict__ w,
                                             const float* __restrict__ bias,
                                             float* __restrict__ ada) {
    int id = blockIdx.x * 256 + threadIdx.x;          // 0 .. 9215
    int j = id % ADA6;
    int b = id / ADA6;
    float s = bias[j];
#pragma unroll 8
    for (int k = 0; k < COND; ++k) s += c[b * COND + k] * w[k * ADA6 + j];
    ada[id] = s;
}

// ---------------------------------------------------------------------------
// 2) fp32 -> bf16 conversion (weights)
// ---------------------------------------------------------------------------
__global__ __launch_bounds__(256) void k_cvt(const float* __restrict__ src,
                                             unsigned short* __restrict__ dst,
                                             int n) {
    int id = blockIdx.x * 256 + threadIdx.x;
    if (id < n) dst[id] = f2bf(src[id]);
}

// ---------------------------------------------------------------------------
// 3) LayerNorm + adaLN modulation -> bf16 activations (one workgroup per row)
// ---------------------------------------------------------------------------
__global__ __launch_bounds__(256) void k_ln_mod(const float* __restrict__ x,
                                                const float* __restrict__ w,
                                                const float* __restrict__ ada,
                                                int shiftOff, int scaleOff,
                                                unsigned short* __restrict__ out) {
    __shared__ float red[256];
    const int row = blockIdx.x;                  // 0..4095
    const int b = row >> 11;
    const float* xr = x + (size_t)row * DIM;
    float s1 = 0.f, s2 = 0.f;
    for (int j = threadIdx.x; j < DIM; j += 256) {
        float v = xr[j]; s1 += v; s2 += v * v;
    }
    red[threadIdx.x] = s1; __syncthreads();
    for (int st = 128; st > 0; st >>= 1) {
        if (threadIdx.x < st) red[threadIdx.x] += red[threadIdx.x + st];
        __syncthreads();
    }
    float mean = red[0] * (1.f / DIM); __syncthreads();
    red[threadIdx.x] = s2; __syncthreads();
    for (int st = 128; st > 0; st >>= 1) {
        if (threadIdx.x < st) red[threadIdx.x] += red[threadIdx.x + st];
        __syncthreads();
    }
    float var = red[0] * (1.f / DIM) - mean * mean;
    float rs = rsqrtf(var + 1e-5f);
    const float* ad = ada + (size_t)b * ADA6;
    for (int j = threadIdx.x; j < DIM; j += 256) {
        float h = (xr[j] - mean) * rs * w[j];
        h = h * (1.f + ad[scaleOff + j]) + ad[shiftOff + j];
        out[(size_t)row * DIM + j] = f2bf(h);
    }
}

// ---------------------------------------------------------------------------
// 4) GEMM: C[M,N] = A[M,K](bf16) @ B[K,N](bf16)
//    block = 256 threads = 8 waves; wave: 16(M) x 64(N); WG tile: 64 x 128.
//    A panel (64x32 = 4KB) staged in LDS via global_load_async_to_lds_b128:
//    exactly one b128 per lane per K-step. B loads are issued between the
//    async issue and the async wait so they overlap the LDS fill.
//    MODE (compile-time): 0 = f32 out; 1 = f32 out + bias;
//                         2 = bias + tanh-GELU -> bf16 out
// ---------------------------------------------------------------------------
template <int MODE>
__global__ __launch_bounds__(256) void k_gemm(const unsigned short* __restrict__ A,
                                              const unsigned short* __restrict__ B,
                                              int K, int N,
                                              const float* __restrict__ bias,
                                              void* __restrict__ Cv) {
    __shared__ unsigned short atile[64 * 32];          // 4 KB A panel
    const int lane = threadIdx.x & 31;
    const int wv = threadIdx.x >> 5;
    const int hi = lane >> 4, mr = lane & 15;
    const int m0 = blockIdx.y * 64 + (wv & 3) * 16;
    const int n0 = blockIdx.x * 128 + (wv >> 2) * 64;

    // per-thread async-copy assignment: row = tid/4 (64 rows), 16B chunk = tid%4
    const int arow = threadIdx.x >> 2;
    const int achk = threadIdx.x & 3;
    const unsigned short* agbase =
        A + (size_t)(blockIdx.y * 64 + arow) * K + achk * 8;
    const unsigned int lds_off =
        (unsigned int)(unsigned long long)(const void*)&atile[arow * 32 + achk * 8];

    f32x8 acc[4] = {};
    for (int k0 = 0; k0 < K; k0 += 32) {
        __syncthreads();   // previous iteration's LDS reads complete
        unsigned long long ga = (unsigned long long)(const void*)(agbase + k0);
        asm volatile("global_load_async_to_lds_b128 %0, %1, off"
                     :: "v"(lds_off), "v"(ga) : "memory");

        // B fragments (global, independent of the LDS fill): lane&15 = N col, elements = K (table order)
        bf16x16 bf[4];
        const __bf16* bcol = reinterpret_cast<const __bf16*>(B + (size_t)k0 * N + n0 + mr);
#pragma unroll
        for (int f = 0; f < 4; ++f)
#pragma unroll
            for (int s = 0; s < 16; ++s) {
                const int kk = (s < 8) ? (8 * hi + s) : (16 + 8 * hi + s - 8);
                bf[f][s] = bcol[(size_t)kk * N + f * 16];
            }

        asm volatile("s_wait_asynccnt 0" ::: "memory");
        __syncthreads();   // all waves' async fills visible

        bf16x16 af = load_a_frag(&atile[((wv & 3) * 16 + mr) * 32], 0, hi);
#pragma unroll
        for (int f = 0; f < 4; ++f)
            acc[f] = __builtin_amdgcn_wmma_f32_16x16x32_bf16(
                false, af, false, bf[f], (short)0, acc[f], false, false);
    }

#pragma unroll
    for (int f = 0; f < 4; ++f) {
        const int n = n0 + f * 16 + mr;
        float bb = 0.f;
        if (MODE != 0) bb = bias[n];
#pragma unroll
        for (int r = 0; r < 8; ++r) {
            const int m = m0 + r + 8 * hi;
            float v = acc[f][r] + bb;
            if (MODE == 2) {
                float t = 0.7978845608028654f * (v + 0.044715f * v * v * v);
                v = 0.5f * v * (1.f + tanhf(t));
                ((unsigned short*)Cv)[(size_t)m * N + n] = f2bf(v);
            } else {
                ((float*)Cv)[(size_t)m * N + n] = v;
            }
        }
    }
}

// ---------------------------------------------------------------------------
// 5) RoPE + scatter to attention layouts:
//    Q [b,h,seq,64] bf16 ; Kt [b,h,64,seq] bf16 ; V [b,h,seq,64] bf16
//    (reference applies RoPE to q, k AND v)
// ---------------------------------------------------------------------------
__global__ __launch_bounds__(256) void k_rope(const float* __restrict__ qkv,
                                              const float* __restrict__ cosb,
                                              const float* __restrict__ sinb,
                                              unsigned short* __restrict__ Q,
                                              unsigned short* __restrict__ Kt,
                                              unsigned short* __restrict__ V) {
    size_t id = (size_t)blockIdx.x * 256 + threadIdx.x;   // 2*2048*2304 exact
    int col = (int)(id % 2304);
    size_t rowi = id / 2304;                 // b*2048 + s
    int s = (int)(rowi & 2047);
    int b = (int)(rowi >> 11);
    int t = col / DIM;                       // 0=q 1=k 2=v
    int hc = col % DIM;
    int head = hc >> 6;
    int d = hc & 63;
    int pos = s & (NHALF - 1);
    float x = qkv[id];
    float xr = (d < 32) ? -qkv[id + 32] : qkv[id - 32];
    float v = x * cosb[pos * HD + d] + xr * sinb[pos * HD + d];
    unsigned short bv = f2bf(v);
    size_t bh = (size_t)(b * HEADS + head);
    if (t == 0)      Q[(bh * SEQ + s) * HD + d] = bv;
    else if (t == 1) Kt[(bh * HD + d) * SEQ + s] = bv;
    else             V[(bh * SEQ + s) * HD + d] = bv;
}

// ---------------------------------------------------------------------------
// 6) Block-sparse flash attention. One wave per 16-query tile.
//    Mask is exactly tile-granular (BLOCK==16): q-block qb attends to
//    count = qb+1 (first half) / qb-63 (second half) dense 16-key tiles.
//    Waves have different trip counts -> no workgroup barriers; P staging
//    is per-wave LDS with an explicit s_wait_dscnt.
// ---------------------------------------------------------------------------
__device__ __forceinline__ int key_tile_seq(int qb, int t) {
    if (qb < 64) return (t == 0) ? qb * 16 : NHALF + (t - 1) * 16;
    return NHALF + t * 16;
}

__global__ __launch_bounds__(128) void k_attn(const unsigned short* __restrict__ Q,
                                              const unsigned short* __restrict__ Kt,
                                              const unsigned short* __restrict__ V,
                                              unsigned short* __restrict__ O) {
    __shared__ unsigned short plds[4][16 * 32];   // per-wave P staging tile
    const int lane = threadIdx.x & 31;
    const int wv = threadIdx.x >> 5;
    const int task = blockIdx.x * 4 + wv;         // 0 .. 3071
    const int qb = task & 127;
    const int bh = task >> 7;                     // b*12 + head
    const int head = bh % HEADS;
    const int b = bh / HEADS;

    const unsigned short* Qbh = Q + (size_t)bh * SEQ * HD;
    const unsigned short* Kbh = Kt + (size_t)bh * HD * SEQ;
    const unsigned short* Vbh = V + (size_t)bh * SEQ * HD;

    const int q0 = (qb < 64) ? qb * 16 : NHALF + (qb - 64) * 16;
    const int count = (qb < 64) ? qb + 1 : qb - 63;

    const int hi = lane >> 4;
    const int mr = lane & 15;

    // Q A-fragments for head-dim chunks [0,32) and [32,64)
    bf16x16 qa[2];
    const unsigned short* qrow = Qbh + (size_t)(q0 + mr) * HD;
    qa[0] = load_a_frag(qrow, 0, hi);
    qa[1] = load_a_frag(qrow, 32, hi);

    float mrun[8], lrun[8];
    f32x8 oacc[4] = {};
#pragma unroll
    for (int r = 0; r < 8; ++r) { mrun[r] = -1e30f; lrun[r] = 0.f; }

    const float scale = 0.125f;                   // 1/sqrt(64)
    const int nchunk = (count + 1) >> 1;
    for (int ch = 0; ch < nchunk; ++ch) {
        const int ta = 2 * ch, tb = 2 * ch + 1;
        const bool hasb = tb < count;
        const int sa = key_tile_seq(qb, ta);
        const int sb = hasb ? key_tile_seq(qb, tb) : sa;

        // batch all K fragments before the score WMMAs (lane&15 = key, elements = head-dim, table order)
        bf16x16 kfa[2], kfb[2];
        const __bf16* Kbf = reinterpret_cast<const __bf16*>(Kbh);
#pragma unroll
        for (int c = 0; c < 2; ++c)
#pragma unroll
            for (int s = 0; s < 16; ++s) {
                const int kk = (s < 8) ? (8 * hi + s) : (16 + 8 * hi + s - 8);
                kfa[c][s] = Kbf[(size_t)(c * 32 + kk) * SEQ + sa + mr];
                kfb[c][s] = Kbf[(size_t)(c * 32 + kk) * SEQ + sb + mr];
            }
        f32x8 Sa = {}, Sb = {};
#pragma unroll
        for (int c = 0; c < 2; ++c) {
            Sa = __builtin_amdgcn_wmma_f32_16x16x32_bf16(
                false, qa[c], false, kfa[c], (short)0, Sa, false, false);
            Sb = __builtin_amdgcn_wmma_f32_16x16x32_bf16(
                false, qa[c], false, kfb[c], (short)0, Sb, false, false);
        }

        float mnew[8];
#pragma unroll
        for (int r = 0; r < 8; ++r) {
            float a = Sa[r] * scale;
            float bv = hasb ? Sb[r] * scale : -1e30f;
            Sa[r] = a; Sb[r] = bv;
            mnew[r] = fmaxf(a, bv);
        }
        // row max across the 16 lanes that share each row
#pragma unroll
        for (int off = 1; off < 16; off <<= 1)
#pragma unroll
            for (int r = 0; r < 8; ++r)
                mnew[r] = fmaxf(mnew[r], __shfl_xor(mnew[r], off, 32));

        float corr[8], psum[8];
#pragma unroll
        for (int r = 0; r < 8; ++r) {
            float mn = fmaxf(mrun[r], mnew[r]);
            corr[r] = __expf(mrun[r] - mn);
            mrun[r] = mn;
            float pa = __expf(Sa[r] - mn);
            float pb = __expf(Sb[r] - mn);
            Sa[r] = pa; Sb[r] = pb;
            psum[r] = pa + pb;
        }
#pragma unroll
        for (int off = 1; off < 16; off <<= 1)
#pragma unroll
            for (int r = 0; r < 8; ++r)
                psum[r] += __shfl_xor(psum[r], off, 32);
#pragma unroll
        for (int r = 0; r < 8; ++r) lrun[r] = lrun[r] * corr[r] + psum[r];
#pragma unroll
        for (int f = 0; f < 4; ++f)
#pragma unroll
            for (int r = 0; r < 8; ++r) oacc[f][r] *= corr[r];

        // P (C/D layout) -> LDS -> A layout for P@V (per-wave, no barrier)
        unsigned short* pw = plds[wv];
#pragma unroll
        for (int r = 0; r < 8; ++r) {
            pw[(r + 8 * hi) * 32 + mr]      = f2bf(Sa[r]);
            pw[(r + 8 * hi) * 32 + 16 + mr] = f2bf(Sb[r]);
        }
        asm volatile("s_wait_dscnt 0" ::: "memory");
        bf16x16 pA = load_a_frag(pw + mr * 32, 0, hi);

        // batch all V fragments: lane&15 = head-dim col, elements = key within the 32-key chunk (table order)
        const __bf16* Vbf = reinterpret_cast<const __bf16*>(Vbh);
        bf16x16 vB[4];
#pragma unroll
        for (int f = 0; f < 4; ++f)
#pragma unroll
            for (int s = 0; s < 16; ++s) {
                const int kk = (s < 8) ? (8 * hi + s) : (16 + 8 * hi + s - 8);
                const int ks = (kk < 16) ? (sa + kk) : (sb + kk - 16);
                vB[f][s] = Vbf[(size_t)ks * HD + f * 16 + mr];
            }
#pragma unroll
        for (int f = 0; f < 4; ++f)
            oacc[f] = __builtin_amdgcn_wmma_f32_16x16x32_bf16(
                false, pA, false, vB[f], (short)0, oacc[f], false, false);
    }

    // normalize + store to o_bf16 [b, seq, 768]
#pragma unroll
    for (int f = 0; f < 4; ++f)
#pragma unroll
        for (int r = 0; r < 8; ++r) {
            float v = oacc[f][r] / lrun[r];
            int srow = q0 + r + 8 * hi;
            int col = head * HD + f * 16 + mr;
            O[((size_t)b * SEQ + srow) * DIM + col] = f2bf(v);
        }
}

// ---------------------------------------------------------------------------
// 7) out = gate * y + skip   (gate = ada[b, gateOff + j])
// ---------------------------------------------------------------------------
__global__ __launch_bounds__(256) void k_gate_add(const float* __restrict__ y,
                                                  const float* __restrict__ skip,
                                                  const float* __restrict__ ada,
                                                  int gateOff,
                                                  float* __restrict__ out) {
    size_t id = (size_t)blockIdx.x * 256 + threadIdx.x;   // 4096*768 exact
    int j = (int)(id % DIM);
    int b = (int)((id / DIM) >> 11);
    out[id] = ada[(size_t)b * ADA6 + gateOff + j] * y[id] + skip[id];
}

// ---------------------------------------------------------------------------
// Host-side orchestration
// ---------------------------------------------------------------------------
extern "C" void kernel_launch(void* const* d_in, const int* in_sizes, int n_in,
                              void* d_out, int out_size, void* d_ws, size_t ws_size,
                              hipStream_t stream) {
    const float* x     = (const float*)d_in[0];
    const float* cosb  = (const float*)d_in[1];
    const float* sinb  = (const float*)d_in[2];
    const float* c     = (const float*)d_in[3];
    const float* Wqkv  = (const float*)d_in[4];
    const float* Wout  = (const float*)d_in[5];
    const float* ln1w  = (const float*)d_in[6];
    const float* ln2w  = (const float*)d_in[7];
    const float* w1    = (const float*)d_in[8];
    const float* b1    = (const float*)d_in[9];
    const float* w2    = (const float*)d_in[10];
    const float* b2    = (const float*)d_in[11];
    const float* adaw  = (const float*)d_in[12];
    const float* adab  = (const float*)d_in[13];
    float* out = (float*)d_out;
    (void)in_sizes; (void)n_in; (void)out_size; (void)ws_size;

    char* p = (char*)d_ws;
    auto alloc = [&](size_t bytes) {
        void* r = (void*)p;
        p += (bytes + 255) & ~(size_t)255;
        return r;
    };
    float*          ada   = (float*)         alloc((size_t)2 * ADA6 * 4);
    unsigned short* wqkvB = (unsigned short*)alloc((size_t)DIM * 3 * DIM * 2);
    unsigned short* woutB = (unsigned short*)alloc((size_t)DIM * DIM * 2);
    unsigned short* w1B   = (unsigned short*)alloc((size_t)DIM * DFF * 2);
    unsigned short* w2B   = (unsigned short*)alloc((size_t)DFF * DIM * 2);
    unsigned short* h1    = (unsigned short*)alloc((size_t)ROWS * DIM * 2);
    float*          qkv   = (float*)         alloc((size_t)ROWS * 3 * DIM * 4);
    unsigned short* Qb    = (unsigned short*)alloc((size_t)2 * HEADS * SEQ * HD * 2);
    unsigned short* Ktb   = (unsigned short*)alloc((size_t)2 * HEADS * HD * SEQ * 2);
    unsigned short* Vb    = (unsigned short*)alloc((size_t)2 * HEADS * SEQ * HD * 2);
    unsigned short* ob    = (unsigned short*)alloc((size_t)ROWS * DIM * 2);
    float*          yb    = (float*)         alloc((size_t)ROWS * DIM * 4);
    float*          x1    = (float*)         alloc((size_t)ROWS * DIM * 4);
    unsigned short* h2    = (unsigned short*)alloc((size_t)ROWS * DIM * 2);
    unsigned short* m1    = (unsigned short*)alloc((size_t)ROWS * DFF * 2);
    float*          m2    = (float*)         alloc((size_t)ROWS * DIM * 4);

    auto cdiv = [](int a, int b) { return (a + b - 1) / b; };

    // weight conversions (deterministic each launch)
    k_cvt<<<cdiv(DIM * 3 * DIM, 256), 256, 0, stream>>>(Wqkv, wqkvB, DIM * 3 * DIM);
    k_cvt<<<cdiv(DIM * DIM, 256),     256, 0, stream>>>(Wout, woutB, DIM * DIM);
    k_cvt<<<cdiv(DIM * DFF, 256),     256, 0, stream>>>(w1, w1B, DIM * DFF);
    k_cvt<<<cdiv(DFF * DIM, 256),     256, 0, stream>>>(w2, w2B, DFF * DIM);

    // adaLN parameters
    k_ada<<<cdiv(2 * ADA6, 256), 256, 0, stream>>>(c, adaw, adab, ada);

    // LN1 + modulation -> h1 (bf16)
    k_ln_mod<<<ROWS, 256, 0, stream>>>(x, ln1w, ada, /*shift*/0, /*scale*/DIM, h1);

    // QKV GEMM: [4096,768] @ [768,2304] -> f32
    k_gemm<0><<<dim3(3 * DIM / 128, ROWS / 64), 256, 0, stream>>>(
        h1, wqkvB, DIM, 3 * DIM, nullptr, (void*)qkv);

    // RoPE + scatter to Q / K^T / V (bf16)
    k_rope<<<(int)(((size_t)ROWS * 3 * DIM) / 256), 256, 0, stream>>>(
        qkv, cosb, sinb, Qb, Ktb, Vb);

    // block-sparse flash attention -> o (bf16, [b,seq,768])
    k_attn<<<(2 * HEADS * 128) / 4, 128, 0, stream>>>(Qb, Ktb, Vb, ob);

    // out projection: [4096,768] @ [768,768] -> f32
    k_gemm<0><<<dim3(DIM / 128, ROWS / 64), 256, 0, stream>>>(
        ob, woutB, DIM, DIM, nullptr, (void*)yb);

    // x1 = gate_msa * y + x
    k_gate_add<<<(int)(((size_t)ROWS * DIM) / 256), 256, 0, stream>>>(
        yb, x, ada, /*gate_msa*/2 * DIM, x1);

    // LN2 + modulation -> h2 (bf16)
    k_ln_mod<<<ROWS, 256, 0, stream>>>(x1, ln2w, ada, /*shift_mlp*/3 * DIM,
                                       /*scale_mlp*/4 * DIM, h2);

    // MLP1: [4096,768] @ [768,3072] + b1, GELU -> bf16
    k_gemm<2><<<dim3(DFF / 128, ROWS / 64), 256, 0, stream>>>(
        h2, w1B, DIM, DFF, b1, (void*)m1);

    // MLP2: [4096,3072] @ [3072,768] + b2 -> f32
    k_gemm<1><<<dim3(DIM / 128, ROWS / 64), 256, 0, stream>>>(
        m1, w2B, DFF, DIM, b2, (void*)m2);

    // out = gate_mlp * mlp + x1
    k_gate_add<<<(int)(((size_t)ROWS * DIM) / 256), 256, 0, stream>>>(
        m2, x1, ada, /*gate_mlp*/5 * DIM, out);
}
